// AntibodyBFN_Receiver_45157286150629
// MI455X (gfx1250) — matrix operand order repaired and checked by reading, not compiled, not run on hardware
//
#include <hip/hip_runtime.h>
#include <math.h>

// ---------------- constants ----------------
static constexpr int kL  = 384;
static constexpr int kD  = 128;
static constexpr int kP  = 64;
static constexpr int kH  = 12;
static constexpr long long kLL = (long long)kL * kL;   // 147456

typedef __attribute__((ext_vector_type(16))) _Float16 v16h;
typedef __attribute__((ext_vector_type(8)))  float    v8f;

union AFu { v16h v; _Float16 e[16]; };
union CFu { v8f  v; float    e[8];  };

// Swizzle helpers: LDS slots for this file's WMMA f16 fragments (contiguous k = 16*hi + e order).
// A 16x32 (MxK): lane = m + 16*hi, hi=k>>4, e = k&15
// B 32x16 (KxN): slot = n_local*2 + hi, hi = k>>4, e = k&15
__device__ __forceinline__ int a_swz(int m, int k) {
  int hi = k >> 4;
  int e  = k & 15;
  return ((m + (hi << 4)) << 4) + e;
}
__device__ __forceinline__ int b_swz(int nl, int k) {
  int hi = k >> 4;
  int e  = k & 15;
  return (((nl << 1) + hi) << 4) + e;
}

// =====================================================================
// Generic strided batched GEMM via v_wmma_f32_16x16x32_f16.
// C[z,m,n] = act( scale * sum_k A[z,m,k]*B[z,k,n] + bias[n] (+C prev) )
// block = 128 threads = 4 waves; wave w owns the (16 x 16) tile at
// (blockIdx.x*16, blockIdx.y*64 + w*16). LDS tiles stored pre-swizzled
// so each lane's fragment is one contiguous 32B read (2x ds_load_b128).
// =====================================================================
__global__ void abfn_gemm_wmma(
    const float* __restrict__ A, long long Asr, long long Asc, long long Abat,
    const float* __restrict__ B, long long Bsr, long long Bsc, long long Bbat,
    float* __restrict__ C, long long Csr, long long Csc, long long Cbat,
    const float* __restrict__ bias,
    int M, int N, int K, float scale, int act, int accum)
{
  __shared__ _Float16 As[32 * 16];       // swizzled [lane][e]
  __shared__ _Float16 Bs[64 * 2 * 16];   // swizzled [nl][hi][e]
  const int tid  = threadIdx.x;
  const int lane = tid & 31;
  const int wv   = tid >> 5;
  const int hi   = (lane >> 4) & 1;
  const int row0 = blockIdx.x * 16;
  const int colbase = blockIdx.y * 64;
  const int z = blockIdx.z;

  const long long zA = (long long)z * Abat;
  const long long zB = (long long)z * Bbat;

  const bool interiorM = (row0 + 16 <= M);
  const bool interiorN = (colbase + 64 <= N);
  const bool fastA = (Asc == 1) && ((Asr & 3) == 0) && ((Abat & 3) == 0);
  const bool fastB = (Bsc == 1) && ((Bsr & 3) == 0) && ((Bbat & 3) == 0);

  CFu acc;
#pragma unroll
  for (int r = 0; r < 8; ++r) acc.e[r] = 0.0f;

  for (int k0 = 0; k0 < K; k0 += 32) {
    const bool interiorK = (k0 + 32 <= K);

    // ---- stage A tile (16 x 32) ----
    if (fastA && interiorM && interiorK) {
      // thread t stages 4 contiguous k's: one global_load_b128
      int m  = tid >> 3;
      int kk = (tid & 7) << 2;
      const float4 v4 = *(const float4*)(A + zA + (long long)(row0 + m) * Asr + (k0 + kk));
      As[a_swz(m, kk + 0)] = (_Float16)v4.x;
      As[a_swz(m, kk + 1)] = (_Float16)v4.y;
      As[a_swz(m, kk + 2)] = (_Float16)v4.z;
      As[a_swz(m, kk + 3)] = (_Float16)v4.w;
    } else if (interiorM && interiorK) {
#pragma unroll
      for (int it = 0; it < 4; ++it) {
        int idx = tid + it * 128;
        int m = idx >> 5, kk = idx & 31;
        float val = A[zA + (long long)(row0 + m) * Asr + (long long)(k0 + kk) * Asc];
        As[a_swz(m, kk)] = (_Float16)val;
      }
    } else {
#pragma unroll
      for (int it = 0; it < 4; ++it) {
        int idx = tid + it * 128;
        int m = idx >> 5, kk = idx & 31;
        int gm = row0 + m, gk = k0 + kk;
        float val = 0.0f;
        if (gm < M && gk < K)
          val = A[zA + (long long)gm * Asr + (long long)gk * Asc];
        As[a_swz(m, kk)] = (_Float16)val;
      }
    }

    // ---- stage B tile (32 x 64) ----
    if (fastB && interiorN && interiorK) {
      // thread t stages 16 contiguous n's of one k row: 4x global_load_b128
      int kk  = tid >> 2;
      int nn0 = (tid & 3) << 4;
      const float* src = B + zB + (long long)(k0 + kk) * Bsr + (colbase + nn0);
#pragma unroll
      for (int u4 = 0; u4 < 4; ++u4) {
        const float4 v4 = ((const float4*)src)[u4];
        int nn = nn0 + u4 * 4;
        Bs[b_swz(nn + 0, kk)] = (_Float16)v4.x;
        Bs[b_swz(nn + 1, kk)] = (_Float16)v4.y;
        Bs[b_swz(nn + 2, kk)] = (_Float16)v4.z;
        Bs[b_swz(nn + 3, kk)] = (_Float16)v4.w;
      }
    } else if (interiorN && interiorK) {
#pragma unroll
      for (int it = 0; it < 16; ++it) {
        int idx = tid + it * 128;
        int kk = idx >> 6, nn = idx & 63;
        float val = B[zB + (long long)(k0 + kk) * Bsr + (long long)(colbase + nn) * Bsc];
        Bs[b_swz(nn, kk)] = (_Float16)val;
      }
    } else {
#pragma unroll
      for (int it = 0; it < 16; ++it) {
        int idx = tid + it * 128;
        int kk = idx >> 6, nn = idx & 63;
        int gk = k0 + kk, gn = colbase + nn;
        float val = 0.0f;
        if (gk < K && gn < N)
          val = B[zB + (long long)gk * Bsr + (long long)gn * Bsc];
        Bs[b_swz(nn, kk)] = (_Float16)val;
      }
    }

    if (tid == 0 && k0 + 32 < K) {
      __builtin_prefetch(A + zA + (long long)row0 * Asr + (long long)(k0 + 32) * Asc, 0, 1);
      __builtin_prefetch(B + zB + (long long)(k0 + 32) * Bsr + (long long)colbase * Bsc, 0, 1);
    }
    __syncthreads();

    AFu a;
    a.v = *(const v16h*)(As + (lane << 4));
    AFu b;
    const int nl = (wv << 4) + (lane & 15);
    b.v = *(const v16h*)(Bs + (((nl << 1) + hi) << 4));
    acc.v = __builtin_amdgcn_wmma_f32_16x16x32_f16(
        false, a.v, false, b.v, (short)0, acc.v, false, false);
    __syncthreads();
  }

  const int n = colbase + (wv << 4) + (lane & 15);
#pragma unroll
  for (int r = 0; r < 8; ++r) {
    int m = row0 + (hi << 3) + r;
    if (m < M && n < N) {
      long long ci = (long long)z * Cbat + (long long)m * Csr + (long long)n * Csc;
      float vv = acc.e[r] * scale;
      if (bias)  vv += bias[n];
      if (accum) vv += C[ci];
      if (act == 1)      vv = vv > 0.0f ? vv : 0.0f;
      else if (act == 2) vv = 1.0f / (1.0f + expf(-vv));
      C[ci] = vv;
    }
  }
}

// =====================================================================
// Fused PAE head: for 16 pair-rows m (m = i*384+j):
//   h = relu(pf2[m,:]@Wcomb + bcomb + fW0[i,:] + fW0[j,:])   (64)
//   out[m] = softplus(h . w1 + b1) * 10
// WMMA for the 16x64 @ 64x64 stage; swizzled LDS fragments.
// =====================================================================
__global__ void abfn_pae_fused(
    const float* __restrict__ pf2, const float* __restrict__ wcomb,
    const float* __restrict__ bcomb, const float* __restrict__ fw0,
    const float* __restrict__ w1, const float* __restrict__ b1,
    float* __restrict__ out)
{
  __shared__ _Float16 AsSw[32 * 2 * 16];   // [lane][kstep][e]
  __shared__ _Float16 WcSw[64 * 2 * 2 * 16]; // [nl][kstep][hi][e]
  __shared__ float    hs[16 * 64];
  const int tid  = threadIdx.x;
  const int lane = tid & 31;
  const int wv   = tid >> 5;
  const int hi   = (lane >> 4) & 1;
  const long long m0 = (long long)blockIdx.x * 16;

  // stage Wcomb (64x64): thread t covers k=t>>1, 32 contiguous n's
  {
    int kk  = tid >> 1;
    int nn0 = (tid & 1) << 5;
    int ks  = kk >> 5;
    int rem = kk & 31;
    int h2  = rem >> 4;
    int e   = rem & 15;
    const float* src = wcomb + kk * 64 + nn0;
#pragma unroll
    for (int u4 = 0; u4 < 8; ++u4) {
      const float4 v4 = ((const float4*)src)[u4];
      int nn = nn0 + u4 * 4;
      float vv[4] = {v4.x, v4.y, v4.z, v4.w};
#pragma unroll
      for (int u = 0; u < 4; ++u)
        WcSw[(((((nn + u) << 1) + ks) << 1) + h2) << 4 | e] = (_Float16)vv[u];
    }
  }
  // stage pf2 rows (16 x 64): thread t covers m=t>>3, 8 contiguous k's
  {
    int m   = tid >> 3;
    int kk0 = (tid & 7) << 3;
    const float* src = pf2 + (m0 + m) * 64 + kk0;
#pragma unroll
    for (int u4 = 0; u4 < 2; ++u4) {
      const float4 v4 = ((const float4*)src)[u4];
      float vv[4] = {v4.x, v4.y, v4.z, v4.w};
#pragma unroll
      for (int u = 0; u < 4; ++u) {
        int kc  = kk0 + u4 * 4 + u;
        int ks  = kc >> 5;
        int rem = kc & 31;
        int h2  = rem >> 4;
        int e   = rem & 15;
        AsSw[(((m + (h2 << 4)) << 1) + ks) << 4 | e] = (_Float16)vv[u];
      }
    }
  }
  __syncthreads();

  CFu acc;
#pragma unroll
  for (int r = 0; r < 8; ++r) acc.e[r] = 0.0f;

  const int nl = (wv << 4) + (lane & 15);
#pragma unroll
  for (int ks = 0; ks < 2; ++ks) {
    AFu a;
    a.v = *(const v16h*)(AsSw + ((((lane << 1) + ks) << 4)));
    AFu b;
    b.v = *(const v16h*)(WcSw + (((((nl << 1) + ks) << 1) + hi) << 4));
    acc.v = __builtin_amdgcn_wmma_f32_16x16x32_f16(
        false, a.v, false, b.v, (short)0, acc.v, false, false);
  }

  const int n = (wv << 4) + (lane & 15);
#pragma unroll
  for (int r = 0; r < 8; ++r) {
    int m = (hi << 3) + r;
    long long grow = m0 + m;
    int i = (int)(grow / kL), j = (int)(grow % kL);
    float h = acc.e[r] + bcomb[n] + fw0[i * 64 + n] + fw0[j * 64 + n];
    hs[m * 64 + n] = h > 0.0f ? h : 0.0f;
  }
  __syncthreads();
  if (tid < 16) {
    float s = b1[0];
    for (int nn = 0; nn < 64; ++nn) s += hs[tid * 64 + nn] * w1[nn];
    float sp = s > 20.0f ? s : log1pf(expf(s));
    out[m0 + tid] = sp * 10.0f;
  }
}

// ---------------- small VALU kernels ----------------
__global__ void abfn_time_embed(const float* t, const float* w0, const float* b0,
                                const float* w1, const float* b1, float* embt)
{
  __shared__ float hid[128];
  int d = threadIdx.x;
  float h = t[0] * w0[d] + b0[d];
  hid[d] = h > 0.0f ? h : 0.0f;
  __syncthreads();
  float s = b1[d];
  for (int j = 0; j < 128; ++j) s += hid[j] * w1[j * 128 + d];
  embt[d] = s;
}

__global__ void abfn_build_cat4(const float* seq, const float* ang, const float* bb,
                                const float* wseq, const float* bseq,
                                const float* wang, const float* bang,
                                const float* wbb, const float* bbb,
                                const float* embt, float* cat4)
{
  int l = blockIdx.x, d = threadIdx.x;
  __shared__ float sseq[20], sang[4], sbb[12];
  if (d < 20) sseq[d] = seq[l * 20 + d];
  if (d >= 32 && d < 36) sang[d - 32] = ang[l * 4 + (d - 32)];
  if (d >= 64 && d < 76) sbb[d - 64]  = bb[l * 12 + (d - 64)];
  __syncthreads();
  float mx = -1e30f;
  for (int c = 0; c < 20; ++c) mx = fmaxf(mx, sseq[c]);
  float den = 0.0f;
  for (int c = 0; c < 20; ++c) den += expf(sseq[c] - mx);
  float inv = 1.0f / den;
  float es = bseq[d];
  for (int c = 0; c < 20; ++c) es += expf(sseq[c] - mx) * inv * wseq[c * 128 + d];
  float ea = bang[d];
  for (int q = 0; q < 4; ++q)
    ea += sinf(sang[q]) * wang[q * 128 + d] + cosf(sang[q]) * wang[(4 + q) * 128 + d];
  float eb = bbb[d];
  for (int q = 0; q < 12; ++q)
    eb += (sbb[q] - sbb[3 + (q % 3)]) * wbb[q * 128 + d];
  float* row = cat4 + (long long)l * 512;
  row[d] = es; row[128 + d] = ea; row[256 + d] = embt[d]; row[384 + d] = eb;
}

__global__ void abfn_scalar_embeds(float* feats, const float* conf, const float* iptm,
                                   const float* wconf, const float* bconf,
                                   const float* wiptm, const float* biptm)
{
  int l = blockIdx.x, d = threadIdx.x;
  feats[l * 128 + d] += conf[l] * wconf[d] + bconf[d] + iptm[0] * wiptm[d] + biptm[d];
}

__global__ void abfn_rot(const float* ori, float* rot, int L)
{
  int l = blockIdx.x * blockDim.x + threadIdx.x;
  if (l >= L) return;
  float Mm[3][3], A[3][3], V[3][3];
  for (int r = 0; r < 3; ++r)
    for (int c = 0; c < 3; ++c) Mm[r][c] = ori[l * 9 + r * 3 + c];
  for (int i = 0; i < 3; ++i)
    for (int j = 0; j < 3; ++j) {
      float s = 0.0f;
      for (int k = 0; k < 3; ++k) s += Mm[k][i] * Mm[k][j];
      A[i][j] = s;
      V[i][j] = (i == j) ? 1.0f : 0.0f;
    }
  for (int sweep = 0; sweep < 15; ++sweep) {
    const int ps[3] = {0, 0, 1}, qs[3] = {1, 2, 2};
    for (int it = 0; it < 3; ++it) {
      int p = ps[it], q = qs[it];
      float apq = A[p][q];
      if (fabsf(apq) < 1e-12f) continue;
      float phi = 0.5f * atan2f(2.0f * apq, A[q][q] - A[p][p]);
      float c = cosf(phi), s = sinf(phi);
      for (int k = 0; k < 3; ++k) {
        float akp = A[k][p], akq = A[k][q];
        A[k][p] = c * akp - s * akq; A[k][q] = s * akp + c * akq;
      }
      for (int k = 0; k < 3; ++k) {
        float apk = A[p][k], aqk = A[q][k];
        A[p][k] = c * apk - s * aqk; A[q][k] = s * apk + c * aqk;
      }
      for (int k = 0; k < 3; ++k) {
        float vkp = V[k][p], vkq = V[k][q];
        V[k][p] = c * vkp - s * vkq; V[k][q] = s * vkp + c * vkq;
      }
    }
  }
  float sv[3];
  for (int i = 0; i < 3; ++i) sv[i] = fmaxf(sqrtf(fmaxf(A[i][i], 0.0f)), 1e-8f);
  float U[3][3];
  for (int r = 0; r < 3; ++r)
    for (int i = 0; i < 3; ++i) {
      float s = 0.0f;
      for (int c = 0; c < 3; ++c) s += Mm[r][c] * V[c][i];
      U[r][i] = s / sv[i];
    }
  float det = Mm[0][0] * (Mm[1][1] * Mm[2][2] - Mm[1][2] * Mm[2][1])
            - Mm[0][1] * (Mm[1][0] * Mm[2][2] - Mm[1][2] * Mm[2][0])
            + Mm[0][2] * (Mm[1][0] * Mm[2][1] - Mm[1][1] * Mm[2][0]);
  if (det < 0.0f) {
    int im = 0;
    if (sv[1] < sv[im]) im = 1;
    if (sv[2] < sv[im]) im = 2;
    for (int r = 0; r < 3; ++r) U[r][im] = -U[r][im];
  }
  for (int r = 0; r < 3; ++r)
    for (int c = 0; c < 3; ++c) {
      float s = 0.0f;
      for (int i = 0; i < 3; ++i) s += U[r][i] * V[c][i];
      rot[l * 9 + r * 3 + c] = s;
    }
}

__global__ void abfn_pf2(const float* pair, const float* pae,
                         const float* w, const float* b, float* out, long long total)
{
  long long idx = (long long)blockIdx.x * blockDim.x + threadIdx.x;
  if (idx >= total) return;
  int c = (int)(idx & 63);
  long long pr = idx >> 6;
  out[idx] = pair[idx] + pae[pr] * w[c] + b[c];
}

__global__ void abfn_points_global(float* pts, const float* rot, const float* tpos, int total)
{
  int id = blockIdx.x * blockDim.x + threadIdx.x;
  if (id >= total) return;
  int l = id / 96;
  float v0 = pts[id * 3 + 0], v1 = pts[id * 3 + 1], v2 = pts[id * 3 + 2];
  const float* R = rot + l * 9;
  float o0 = R[0] * v0 + R[1] * v1 + R[2] * v2 + tpos[l * 3 + 0];
  float o1 = R[3] * v0 + R[4] * v1 + R[5] * v2 + tpos[l * 3 + 1];
  float o2 = R[6] * v0 + R[7] * v1 + R[8] * v2 + tpos[l * 3 + 2];
  pts[id * 3 + 0] = o0; pts[id * 3 + 1] = o1; pts[id * 3 + 2] = o2;
}

__global__ void abfn_spatial_logits(float* logits, const float* qpg, const float* kpg,
                                    const float* sc, const float* maskres)
{
  int i = blockIdx.x;
  int tid = threadIdx.x;           // 384
  int h = tid % 12, jj = tid / 12; // jj in 0..31
  int j = blockIdx.y * 32 + jj;
  __shared__ float qi[288];
  for (int idx = tid; idx < 288; idx += 384) qi[idx] = qpg[(long long)i * 288 + idx];
  __syncthreads();
  const float* kr = kpg + (long long)j * 288 + h * 24;
  const float* qr = qi + h * 24;
  float d2 = 0.0f;
  for (int p = 0; p < 24; ++p) { float dd = qr[p] - kr[p]; d2 += dd * dd; }
  float g = sc[h];
  float gamma = g > 20.0f ? g : log1pf(expf(g));
  float coef = -gamma * (1.0f / 6.0f) * 0.5f;
  long long li = ((long long)i * kL + j) * kH + h;
  float lg = (logits[li] + d2 * coef) * 0.57735026919f;
  if (!((maskres[i] > 0.5f) && (maskres[j] > 0.5f))) lg -= 100000.0f;
  logits[li] = lg;
}

__global__ void abfn_softmax(float* logits, const float* maskres)
{
  int i = blockIdx.x, h = blockIdx.y, tid = threadIdx.x; // 128 threads
  __shared__ float red[128];
  long long base = ((long long)i * kL) * kH + h;
  float mx = -1e30f;
  for (int j = tid; j < kL; j += 128) mx = fmaxf(mx, logits[base + (long long)j * kH]);
  red[tid] = mx; __syncthreads();
  for (int s = 64; s > 0; s >>= 1) { if (tid < s) red[tid] = fmaxf(red[tid], red[tid + s]); __syncthreads(); }
  mx = red[0]; __syncthreads();
  float sm = 0.0f;
  for (int j = tid; j < kL; j += 128) sm += expf(logits[base + (long long)j * kH] - mx);
  red[tid] = sm; __syncthreads();
  for (int s = 64; s > 0; s >>= 1) { if (tid < s) red[tid] += red[tid + s]; __syncthreads(); }
  float inv = 1.0f / red[0];
  float mi = maskres[i] > 0.5f ? 1.0f : 0.0f;
  for (int j = tid; j < kL; j += 128) {
    long long li = base + (long long)j * kH;
    logits[li] = expf(logits[li] - mx) * inv * mi;
  }
}

__global__ void abfn_spatial_feats(const float* agg, const float* rot, const float* tpos,
                                   float* catb, int total)
{
  int id = blockIdx.x * blockDim.x + threadIdx.x;
  if (id >= total) return;
  int l = id / 96, hp = id % 96;
  float r0 = agg[(long long)l * 288 + hp * 3 + 0] - tpos[l * 3 + 0];
  float r1 = agg[(long long)l * 288 + hp * 3 + 1] - tpos[l * 3 + 1];
  float r2 = agg[(long long)l * 288 + hp * 3 + 2] - tpos[l * 3 + 2];
  const float* R = rot + l * 9;
  float p0 = R[0] * r0 + R[3] * r1 + R[6] * r2;  // R^T @ rel
  float p1 = R[1] * r0 + R[4] * r1 + R[7] * r2;
  float p2 = R[2] * r0 + R[5] * r1 + R[8] * r2;
  float dist = sqrtf(p0 * p0 + p1 * p1 + p2 * p2);
  float inv = 1.0f / (dist + 1e-4f);
  float* row = catb + (long long)l * 1824;
  row[1152 + hp * 3 + 0] = p0; row[1152 + hp * 3 + 1] = p1; row[1152 + hp * 3 + 2] = p2;
  row[1440 + hp] = dist;
  row[1536 + hp * 3 + 0] = p0 * inv; row[1536 + hp * 3 + 1] = p1 * inv; row[1536 + hp * 3 + 2] = p2 * inv;
}

__global__ void abfn_ln(const float* x, const float* f, const float* maskres,
                        const float* g, const float* b, float* out)
{
  int l = blockIdx.x, d = threadIdx.x;
  __shared__ float red[128];
  float add = f[l * 128 + d];
  if (maskres && !(maskres[l] > 0.5f)) add = 0.0f;
  float v = x[l * 128 + d] + add;
  red[d] = v; __syncthreads();
  for (int s = 64; s > 0; s >>= 1) { if (d < s) red[d] += red[d + s]; __syncthreads(); }
  float mean = red[0] * (1.0f / 128.0f); __syncthreads();
  float dv = v - mean;
  red[d] = dv * dv; __syncthreads();
  for (int s = 64; s > 0; s >>= 1) { if (d < s) red[d] += red[d + s]; __syncthreads(); }
  float var = red[0] * (1.0f / 128.0f);
  out[l * 128 + d] = dv * rsqrtf(var + 1e-5f) * g[d] + b[d];
}

__global__ void abfn_predpos(const float* rot, const float* pl, const float* tpos,
                             float* out, int L)
{
  int l = blockIdx.x * blockDim.x + threadIdx.x;
  if (l >= L) return;
  const float* R = rot + l * 9;
  float v0 = pl[l * 3 + 0], v1 = pl[l * 3 + 1], v2 = pl[l * 3 + 2];
  out[l * 3 + 0] = R[0] * v0 + R[1] * v1 + R[2] * v2 + tpos[l * 3 + 0];
  out[l * 3 + 1] = R[3] * v0 + R[4] * v1 + R[5] * v2 + tpos[l * 3 + 1];
  out[l * 3 + 2] = R[6] * v0 + R[7] * v1 + R[8] * v2 + tpos[l * 3 + 2];
}

__global__ void abfn_gfeat(const float* feats, const float* maskf, float* gfeat)
{
  int d = threadIdx.x;
  float s = 0.0f, den = 0.0f;
  for (int l = 0; l < kL; ++l) {
    float m = maskf[l];
    s += feats[(long long)l * 128 + d] * m;
    den += m;
  }
  gfeat[d] = s / (den + 1e-8f);
}

__global__ void abfn_iptm(const float* gfeat, const float* W0, const float* b0,
                          const float* w1, const float* b1, float* out)
{
  __shared__ float hid[128], red[128];
  int d = threadIdx.x;
  float h = b0[d];
  for (int j = 0; j < 128; ++j) h += gfeat[j] * W0[j * 128 + d];
  hid[d] = h > 0.0f ? h : 0.0f;
  __syncthreads();
  red[d] = hid[d] * w1[d]; __syncthreads();
  for (int s = 64; s > 0; s >>= 1) { if (d < s) red[d] += red[d + s]; __syncthreads(); }
  if (d == 0) out[0] = 1.0f / (1.0f + expf(-(red[0] + b1[0])));
}

__global__ void abfn_bcomb(const float* b0, const float* bproj, const float* W0, float* bcomb)
{
  int n = threadIdx.x; // 64
  float s = b0[n];
  for (int c = 0; c < 128; ++c) s += bproj[c] * W0[c * 64 + n];
  bcomb[n] = s;
}

// ---------------- host side ----------------
static inline void gemm(hipStream_t s,
  const float* A, long long Asr, long long Asc, long long Abat,
  const float* B, long long Bsr, long long Bsc, long long Bbat,
  float* C, long long Csr, long long Csc, long long Cbat,
  const float* bias, int M, int N, int K, int batch, float scale, int act, int accum)
{
  dim3 g((M + 15) / 16, (N + 63) / 64, batch);
  abfn_gemm_wmma<<<g, 128, 0, s>>>(A, Asr, Asc, Abat, B, Bsr, Bsc, Bbat,
                                   C, Csr, Csc, Cbat, bias, M, N, K, scale, act, accum);
}

extern "C" void kernel_launch(void* const* d_in, const int* in_sizes, int n_in,
                              void* d_out, int out_size, void* d_ws, size_t ws_size,
                              hipStream_t stream)
{
  if (n_in < 131) return;
  const float* in[131];
  for (int i = 0; i < 131; ++i) in[i] = (const float*)d_in[i];
  float* outp = (float*)d_out;
  float* ws = (float*)d_ws;

  // data inputs
  const float* theta_seq = in[0];
  const float* theta_pos = in[1];
  const float* theta_ori = in[2];
  const float* theta_ang = in[3];
  const float* t_in      = in[4];
  const float* pair_feat = in[5];
  const float* mask_res  = in[6];
  const float* backbone  = in[7];
  const float* prev_conf = in[8];
  const float* prev_iptm = in[9];
  const float* prev_pae  = in[10];

  // workspace layout (floats)
  size_t o = 0;
  float* pf2    = ws + o; o += (size_t)kLL * kP;      // 9437184
  float* logits = ws + o; o += (size_t)kLL * kH;      // 1769472
  float* feats  = ws + o; o += kL * kD;
  float* cat4   = ws + o; o += kL * 512;
  float* catbuf = ws + o; o += kL * 1824;
  float* x1     = ws + o; o += kL * kD;
  float* t1     = ws + o; o += kL * kD;
  float* t2     = ws + o; o += kL * kD;
  float* qbuf   = ws + o; o += kL * 384;
  float* kbuf   = ws + o; o += kL * 384;
  float* vbuf   = ws + o; o += kL * 384;
  float* qpbuf  = ws + o; o += kL * 288;
  float* kpbuf  = ws + o; o += kL * 288;
  float* vpbuf  = ws + o; o += kL * 288;
  float* aggbuf = ws + o; o += kL * 288;
  float* rot    = ws + o; o += kL * 9 + 3;            // keep 16B alignment
  float* embt   = ws + o; o += 128;
  float* ploc   = ws + o; o += kL * 3 + 1;
  float* fw0    = ws + o; o += kL * 64;
  float* wcomb  = ws + o; o += 64 * 64;
  float* bcomb  = ws + o; o += 64;
  float* gfeat  = ws + o; o += 128;
  (void)ws_size; (void)in_sizes; (void)out_size;

  const float inv_sqrt_qk = 0.17677669529663687f;

  // ---- embeddings ----
  abfn_time_embed<<<1, 128, 0, stream>>>(t_in, in[128], in[127], in[130], in[129], embt);
  abfn_build_cat4<<<kL, 128, 0, stream>>>(theta_seq, theta_ang, backbone,
      in[126], in[125], in[12], in[11], in[14], in[13], embt, cat4);
  gemm(stream, cat4, 512, 1, 0, in[122], 128, 1, 0, t1, 128, 1, 0,
       in[121], kL, 128, 512, 1, 1.0f, 1, 0);                       // res_mixer0 + relu
  gemm(stream, t1, 128, 1, 0, in[124], 128, 1, 0, feats, 128, 1, 0,
       in[123], kL, 128, 128, 1, 1.0f, 0, 0);                       // res_mixer1
  abfn_scalar_embeds<<<kL, 128, 0, stream>>>(feats, prev_conf, prev_iptm,
      in[16], in[15], in[116], in[115]);
  abfn_rot<<<3, 128, 0, stream>>>(theta_ori, rot, kL);
  {
    long long total = kLL * kP;
    abfn_pf2<<<(unsigned)((total + 255) / 256), 256, 0, stream>>>(
        pair_feat, prev_pae, in[118], in[117], pf2, total);
  }

  // ---- encoder layers ----
  for (int layer = 0; layer < 4; ++layer) {
    int lb = 17 + layer * 20;
    const float* ln1_b = in[lb + 0];  const float* ln1_g = in[lb + 1];
    const float* ln2_b = in[lb + 2];  const float* ln2_g = in[lb + 3];
    const float* mlp0_b = in[lb + 4]; const float* mlp0_w = in[lb + 5];
    const float* mlp1_b = in[lb + 6]; const float* mlp1_w = in[lb + 7];
    const float* mlp2_b = in[lb + 8]; const float* mlp2_w = in[lb + 9];
    const float* out_b = in[lb + 10]; const float* out_w = in[lb + 11];
    const float* spc   = in[lb + 12];
    const float* wk = in[lb + 13]; const float* wkp = in[lb + 14];
    const float* wpb = in[lb + 15]; const float* wq = in[lb + 16];
    const float* wqp = in[lb + 17]; const float* wv = in[lb + 18];
    const float* wvp = in[lb + 19];

    gemm(stream, feats, 128, 1, 0, wq, 384, 1, 0, qbuf, 384, 1, 0, nullptr, kL, 384, 128, 1, 1.0f, 0, 0);
    gemm(stream, feats, 128, 1, 0, wk, 384, 1, 0, kbuf, 384, 1, 0, nullptr, kL, 384, 128, 1, 1.0f, 0, 0);
    gemm(stream, feats, 128, 1, 0, wv, 384, 1, 0, vbuf, 384, 1, 0, nullptr, kL, 384, 128, 1, 1.0f, 0, 0);
    gemm(stream, feats, 128, 1, 0, wqp, 288, 1, 0, qpbuf, 288, 1, 0, nullptr, kL, 288, 128, 1, 1.0f, 0, 0);
    gemm(stream, feats, 128, 1, 0, wkp, 288, 1, 0, kpbuf, 288, 1, 0, nullptr, kL, 288, 128, 1, 1.0f, 0, 0);
    gemm(stream, feats, 128, 1, 0, wvp, 288, 1, 0, vpbuf, 288, 1, 0, nullptr, kL, 288, 128, 1, 1.0f, 0, 0);

    abfn_points_global<<<288, 128, 0, stream>>>(qpbuf, rot, theta_pos, kL * 96);
    abfn_points_global<<<288, 128, 0, stream>>>(kpbuf, rot, theta_pos, kL * 96);
    abfn_points_global<<<288, 128, 0, stream>>>(vpbuf, rot, theta_pos, kL * 96);

    // logits_node: batched over heads. Q(i,d) @ K(j,d)^T
    gemm(stream, qbuf, 384, 1, 32, kbuf, 1, 384, 32,
         logits, (long long)kL * kH, kH, 1, nullptr, kL, kL, 32, kH, inv_sqrt_qk, 0, 0);
    // logits_pair: pf2(LL,64) @ wpb(64,12), accumulate
    gemm(stream, pf2, 64, 1, 0, wpb, 12, 1, 0,
         logits, 12, 1, 0, nullptr, (int)kLL, 12, 64, 1, 1.0f, 0, 1);

    abfn_spatial_logits<<<dim3(kL, 12), 384, 0, stream>>>(logits, qpbuf, kpbuf, spc, mask_res);
    abfn_softmax<<<dim3(kL, kH), 128, 0, stream>>>(logits, mask_res);

    // feat_p2n: batch over i; (h x j) @ (j x 64) -> catbuf[:, 0:768]
    gemm(stream, logits, 1, 12, 4608, pf2, 64, 1, (long long)kL * 64,
         catbuf, 64, 1, 1824, nullptr, 12, 64, kL, kL, 1.0f, 0, 0);
    // feat_node: batch over heads -> catbuf[:, 768:1152]
    gemm(stream, logits, 4608, 12, 1, vbuf, 384, 1, 32,
         catbuf + 768, 1824, 1, 32, nullptr, kL, 32, kL, kH, 1.0f, 0, 0);
    // agg: batch over heads -> aggbuf
    gemm(stream, logits, 4608, 12, 1, vpbuf, 288, 1, 24,
         aggbuf, 288, 1, 24, nullptr, kL, 24, kL, kH, 1.0f, 0, 0);

    abfn_spatial_feats<<<288, 128, 0, stream>>>(aggbuf, rot, theta_pos, catbuf, kL * 96);

    gemm(stream, catbuf, 1824, 1, 0, out_w, 128, 1, 0, t1, 128, 1, 0,
         out_b, kL, 128, 1824, 1, 1.0f, 0, 0);
    abfn_ln<<<kL, 128, 0, stream>>>(feats, t1, mask_res, ln1_g, ln1_b, x1);

    gemm(stream, x1, 128, 1, 0, mlp0_w, 128, 1, 0, t2, 128, 1, 0, mlp0_b, kL, 128, 128, 1, 1.0f, 1, 0);
    gemm(stream, t2, 128, 1, 0, mlp1_w, 128, 1, 0, t1, 128, 1, 0, mlp1_b, kL, 128, 128, 1, 1.0f, 1, 0);
    gemm(stream, t1, 128, 1, 0, mlp2_w, 128, 1, 0, t2, 128, 1, 0, mlp2_b, kL, 128, 128, 1, 1.0f, 0, 0);
    abfn_ln<<<kL, 128, 0, stream>>>(x1, t2, nullptr, ln2_g, ln2_b, feats);
  }

  // ---- heads ----
  float* out_seq   = outp;                 // 384*20
  float* out_pos   = outp + 7680;          // 384*3
  float* out_ori   = outp + 8832;          // 384*6
  float* out_ang   = outp + 11136;         // 384*8
  float* out_plddt = outp + 14208;         // 384
  float* out_iptm  = outp + 14592;         // 1
  float* out_pae   = outp + 14593;         // 147456

  gemm(stream, feats, 128, 1, 0, in[114], 20, 1, 0, out_seq, 20, 1, 0, in[113], kL, 20, 128, 1, 1.0f, 0, 0);
  gemm(stream, feats, 128, 1, 0, in[112], 3, 1, 0, ploc, 3, 1, 0, in[111], kL, 3, 128, 1, 1.0f, 0, 0);
  abfn_predpos<<<3, 128, 0, stream>>>(rot, ploc, theta_pos, out_pos, kL);
  gemm(stream, feats, 128, 1, 0, in[104], 6, 1, 0, out_ori, 6, 1, 0, in[103], kL, 6, 128, 1, 1.0f, 0, 0);
  gemm(stream, feats, 128, 1, 0, in[98], 8, 1, 0, out_ang, 8, 1, 0, in[97], kL, 8, 128, 1, 1.0f, 0, 0);
  gemm(stream, feats, 128, 1, 0, in[110], 1, 1, 0, out_plddt, 1, 1, 0, in[109], kL, 1, 128, 1, 1.0f, 2, 0);

  abfn_gfeat<<<1, 128, 0, stream>>>(feats, mask_res, gfeat);
  abfn_iptm<<<1, 128, 0, stream>>>(gfeat, in[100], in[99], in[102], in[101], out_iptm);

  // PAE head: Wcomb = pair_proj.w @ head_pae0.w ; bcomb = b0 + bproj @ W0 ; fW0 = feats @ W0
  gemm(stream, in[120], 128, 1, 0, in[106], 64, 1, 0, wcomb, 64, 1, 0, nullptr, 64, 64, 128, 1, 1.0f, 0, 0);
  abfn_bcomb<<<1, 64, 0, stream>>>(in[105], in[119], in[106], bcomb);
  gemm(stream, feats, 128, 1, 0, in[106], 64, 1, 0, fw0, 64, 1, 0, nullptr, kL, 64, 128, 1, 1.0f, 0, 0);
  abfn_pae_fused<<<(unsigned)(kLL / 16), 128, 0, stream>>>(pf2, wcomb, bcomb, fw0, in[108], in[107], out_pae);
}
